// VRNN_28802050687063
// MI455X (gfx1250) — matrix operand order repaired and checked by reading, not compiled, not run on hardware
//
#include <hip/hip_runtime.h>

typedef _Float16 f16;
typedef _Float16 v16h __attribute__((ext_vector_type(16)));
typedef _Float16 v8h  __attribute__((ext_vector_type(8)));
typedef _Float16 v4h  __attribute__((ext_vector_type(4)));
typedef float    v8f  __attribute__((ext_vector_type(8)));
typedef float    v4f  __attribute__((ext_vector_type(4)));
typedef int      vsi4 __attribute__((vector_size(16)));  // int4, matches builtin proto

// gfx1250 async global->LDS DMA path (ASYNCcnt), if the toolchain exposes it
#if __has_builtin(__builtin_amdgcn_global_load_async_to_lds_b128)
#define HAVE_ASYNC_LDS 1
#else
#define HAVE_ASYNC_LDS 0
#endif

static __device__ __forceinline__ void wait_async0() {
#if __has_builtin(__builtin_amdgcn_s_wait_asynccnt)
  __builtin_amdgcn_s_wait_asynccnt(0);
#else
  asm volatile("s_wait_asynccnt 0x0" ::: "memory");
#endif
}

#if HAVE_ASYNC_LDS
static __device__ __forceinline__ void async_b128(const float* g, float* l) {
  __builtin_amdgcn_global_load_async_to_lds_b128(
      (__attribute__((address_space(1))) vsi4*)g,
      (__attribute__((address_space(3))) vsi4*)l, 0, 0);
}
#endif

// ---------------- problem constants ----------------
constexpr int B = 4096, T = 200, P = 32, H = 128, Z = 64;
constexpr int WAVES = 4;                 // waves per workgroup
constexpr int NWG   = B / (16 * WAVES);  // 64 workgroups, 256 wave-tiles

// ---------------- packed f16 weight layout (units: halfs) ----------------
// Each 32(K)x16(N) B fragment = 512 halfs, lane-major: lane*16 + e
//   lanes 0-15:  n = lane,    k = kt*32 + e       (e = 0..15)
//   lanes 16-31: n = lane-16, k = kt*32 + 16 + e
// Fragment index within a layer: frag = nt*KT_total + kt  (kt contiguous)
constexpr int FR = 512;
constexpr int OFF_EMBX1 = 0;                       // K32 x N128 : 1x8
constexpr int OFF_EMBX2 = OFF_EMBX1 +   8 * FR;    // K128x N64  : 4x4
constexpr int OFF_ENC1  = OFF_EMBX2 +  16 * FR;    // K192x N128 : 6x8
constexpr int OFF_ENC2  = OFF_ENC1  +  48 * FR;    // K128x N128 : 4x8
constexpr int OFF_PRI1  = OFF_ENC2  +  32 * FR;    // K128x N128 : 4x8
constexpr int OFF_PRI2  = OFF_PRI1  +  32 * FR;    // K128x N128 : 4x8
constexpr int OFF_EMBZ1 = OFF_PRI2  +  32 * FR;    // K64 x N128 : 2x8
constexpr int OFF_EMBZ2 = OFF_EMBZ1 +  16 * FR;    // K128x N64  : 4x4
constexpr int OFF_DEC1  = OFF_EMBZ2 +  16 * FR;    // K192x N128 : 6x8
constexpr int OFF_DEC2  = OFF_DEC1  +  48 * FR;    // K128x N32  : 4x2
constexpr int OFF_LSTM  = OFF_DEC2  +   8 * FR;    // K256x N512 : 8x32 (W_ih kt0-3, W_hh kt4-7)
constexpr int PACKED_TOTAL = OFF_LSTM + 256 * FR;  // 262144 halfs = 512 KB

// ---------------- per-wave LDS slice (bytes) ----------------
constexpr int LDS_XT  = 0;       // f16[16][32]   1024
constexpr int LDS_EP  = 1024;    // f32[16][64]   4096
constexpr int LDS_H   = 5120;    // f16[16][128]  4096
constexpr int LDS_HID = 9216;    // f16[16][128]  4096
constexpr int LDS_EX  = 13312;   // f16[16][64]   2048
constexpr int LDS_Z   = 15360;   // f16[16][64]   2048
constexpr int LDS_EZ  = 17408;   // f16[16][64]   2048
constexpr int LDS_EXH = 19456;   // f16[16][64]   2048
constexpr int LDS_XH  = 21504;   // f16[16][32]   1024
constexpr int LDS_C   = 22528;   // f32[16][128]  8192  (LSTM cell state)
constexpr int LDS_PER_WAVE = 30720;                // 30 KB * 4 = 120 KB/WG

// ---------------- WMMA helpers ----------------
static __device__ __forceinline__ v8f wmma16(v16h a, v16h b, v8f c) {
  // D(16x16 f32) = A(16x32 f16) * B(32x16 f16) + C
  return __builtin_amdgcn_wmma_f32_16x16x32_f16(false, a, false, b, (short)0, c,
                                                false, false);
}

// A fragment (16x32 f16) gathered from row-major LDS buffer [16][rowlen]
// lanes 0-15: m=lane, K {0..15}; lanes 16-31: m=lane-16, K {16..31}
static __device__ __forceinline__ v16h load_A16(const f16* buf, int rowlen,
                                                int kt, int lane) {
  int m  = lane & 15;
  int kb = kt * 32 + ((lane >> 4) << 4);
  const f16* p = buf + m * rowlen + kb;
  v8h lo = *(const v8h*)(p);       // K kb..kb+7
  v8h hi = *(const v8h*)(p + 8);   // K kb+8..kb+15
  return __builtin_shufflevector(lo, hi, 0, 1, 2, 3, 4, 5, 6, 7, 8, 9, 10, 11,
                                 12, 13, 14, 15);
}

// B fragment from packed global weights: 32B contiguous per lane (2x b128)
static __device__ __forceinline__ v16h load_B16(const f16* wp, int frag,
                                                int lane) {
  const f16* p = wp + ((size_t)frag << 9) + (lane << 4);
  v8h lo = *(const v8h*)(p);
  v8h hi = *(const v8h*)(p + 8);
  return __builtin_shufflevector(lo, hi, 0, 1, 2, 3, 4, 5, 6, 7, 8, 9, 10, 11,
                                 12, 13, 14, 15);
}

// C init = bias broadcast (all 8 rows of a lane share column n)
static __device__ __forceinline__ v8f bias_frag(const float* b, int nt,
                                                int lane) {
  float x = b[nt * 16 + (lane & 15)];
  v8f c;
#pragma unroll
  for (int r = 0; r < 8; ++r) c[r] = x;
  return c;
}
static __device__ __forceinline__ v8f bias2_frag(const float* b1,
                                                 const float* b2, int nt,
                                                 int lane) {
  int i = nt * 16 + (lane & 15);
  float x = b1[i] + b2[i];
  v8f c;
#pragma unroll
  for (int r = 0; r < 8; ++r) c[r] = x;
  return c;
}

static __device__ __forceinline__ float relu1(float v) {
  // single v_med3_f32: med3(v, 0, +big) == relu(v)
  return __builtin_amdgcn_fmed3f(v, 0.0f, 3.0e38f);
}

// D fragment -> (ReLU) -> f16 row-major LDS buffer
static __device__ __forceinline__ void store_frag(v8f d, f16* buf, int rowlen,
                                                  int nt, int lane, bool relu) {
  int col = nt * 16 + (lane & 15);
  int mb  = (lane >> 4) << 3;
#pragma unroll
  for (int r = 0; r < 8; ++r) {
    float v = d[r];
    if (relu) v = relu1(v);
    buf[(mb + r) * rowlen + col] = (f16)v;
  }
}

// one dense layer: runtime loop over N tiles keeps code small + regs bounded
template <int NK, int NN>
static __device__ __forceinline__ void mlp_gemm(const v16h* A, const f16* wp,
                                                const float* bias, f16* obuf,
                                                int orow, int lane, bool relu) {
#pragma unroll 1
  for (int nt = 0; nt < NN; ++nt) {
    v8f acc = bias_frag(bias, nt, lane);
#pragma unroll
    for (int k = 0; k < NK; ++k)
      acc = wmma16(A[k], load_B16(wp, nt * NK + k, lane), acc);
    store_frag(acc, obuf, orow, nt, lane, relu);
  }
}

static __device__ __forceinline__ float sigmf(float x) {
  return 1.f / (1.f + __expf(-x));
}
static __device__ __forceinline__ float tanhfast(float x) {
  // tanh(x) = 1 - 2/(exp(2x)+1)   (one v_exp + v_rcp)
  float e = __expf(2.f * x);
  return 1.f - 2.f / (e + 1.f);
}

// ---------------- weight packing (f32 row-major [K][N] -> f16 fragments) ----
__global__ void pack_kernel(const float* __restrict__ W, int N, int KT_src,
                            int NT, int kt_base, int KT_total,
                            f16* __restrict__ dst) {
  int total = KT_src * NT * FR;
  for (int gid = blockIdx.x * blockDim.x + threadIdx.x; gid < total;
       gid += gridDim.x * blockDim.x) {
    int frag = gid >> 9;
    int r    = gid & 511;
    int lane = r >> 4, e = r & 15;
    int kt = frag % KT_src, nt = frag / KT_src;
    int n = nt * 16 + (lane & 15);
    int k = kt * 32 + ((lane >> 4) << 4) + e;
    dst[((size_t)(nt * KT_total + kt_base + kt) << 9) + (lane << 4) + e] =
        (f16)W[(size_t)k * N + n];
  }
}

// ---------------- persistent VRNN scan kernel ----------------
__global__ __launch_bounds__(WAVES * 32) void vrnn_kernel(
    const float* __restrict__ x, const float* __restrict__ eps,
    const f16* __restrict__ pk,
    const float* __restrict__ embx_b1, const float* __restrict__ embx_b2,
    const float* __restrict__ enc_b1,  const float* __restrict__ enc_b2,
    const float* __restrict__ prior_b1,const float* __restrict__ prior_b2,
    const float* __restrict__ embz_b1, const float* __restrict__ embz_b2,
    const float* __restrict__ dec_b1,  const float* __restrict__ dec_b2,
    const float* __restrict__ b_ih,    const float* __restrict__ b_hh,
    float* __restrict__ out) {
  const int lane = threadIdx.x & 31;
  const int wave = threadIdx.x >> 5;
  const int b0   = (blockIdx.x * WAVES + wave) * 16;

  __shared__ __align__(16) unsigned char smem[WAVES * LDS_PER_WAVE];
  unsigned char* wls = smem + wave * LDS_PER_WAVE;
  f16*   s_xt  = (f16*)(wls + LDS_XT);
  float* s_ep  = (float*)(wls + LDS_EP);
  f16*   s_h   = (f16*)(wls + LDS_H);
  f16*   s_hid = (f16*)(wls + LDS_HID);
  f16*   s_ex  = (f16*)(wls + LDS_EX);
  f16*   s_z   = (f16*)(wls + LDS_Z);
  f16*   s_ez  = (f16*)(wls + LDS_EZ);
  f16*   s_exh = (f16*)(wls + LDS_EXH);
  f16*   s_xh  = (f16*)(wls + LDS_XH);
  float* s_c   = (float*)(wls + LDS_C);

  float* out_rec = out;                        // [B,T,P]
  float* out_zp  = out + (size_t)B * T * P;    // [B,T,2,Z]
  float* out_pp  = out_zp + (size_t)B * T * 2 * Z;

  // h0 = 0, c0 = 0
  for (int i = lane; i < 16 * H; i += 32) s_h[i] = (f16)0.f;
  for (int i = lane; i < 16 * H; i += 32) s_c[i] = 0.f;

  const int colb = lane & 15;        // n within tile
  const int mb   = (lane >> 4) << 3; // row base for D fragments

#pragma unroll 1
  for (int t = 0; t < T; ++t) {
    // ---- eps_t: async DMA global -> LDS (ASYNCcnt), raw f32 copy ----
#if HAVE_ASYNC_LDS
#pragma unroll
    for (int i = 0; i < 8; ++i) {
      int idx = i * 32 + lane;
      int m = idx >> 4, c4 = (idx & 15) * 4;
      async_b128(eps + ((size_t)t * B + b0 + m) * Z + c4, s_ep + m * Z + c4);
    }
#else
#pragma unroll 1
    for (int idx = lane; idx < 16 * 16; idx += 32) {
      int m = idx >> 4, c4 = (idx & 15) * 4;
      *(v4f*)(s_ep + m * Z + c4) =
          *(const v4f*)(eps + ((size_t)t * B + b0 + m) * Z + c4);
    }
#endif

    // ---- stage x_t (f32->f16) into LDS; prefetch t+1 ----
#pragma unroll 1
    for (int idx = lane; idx < 16 * 8; idx += 32) {
      int m = idx >> 3, c4 = (idx & 7) * 4;
      v4f v = *(const v4f*)(x + ((size_t)(b0 + m) * T + t) * P + c4);
      v4h h4;
#pragma unroll
      for (int r = 0; r < 4; ++r) h4[r] = (f16)v[r];
      *(v4h*)(s_xt + m * P + c4) = h4;
    }
    if (t + 1 < T) {
      int m = lane & 15;
      __builtin_prefetch(x + ((size_t)(b0 + m) * T + (t + 1)) * P, 0, 1);
      __builtin_prefetch(eps + ((size_t)(t + 1) * B + b0 + m) * Z, 0, 1);
    }

    v16h A[6];

    // ---- ex = embx MLP(x_t): 32 -> 128 relu -> 64 ----
    A[0] = load_A16(s_xt, P, 0, lane);
    mlp_gemm<1, 8>(A, pk + OFF_EMBX1, embx_b1, s_hid, H, lane, true);
#pragma unroll
    for (int k = 0; k < 4; ++k) A[k] = load_A16(s_hid, H, k, lane);
    mlp_gemm<4, 4>(A, pk + OFF_EMBX2, embx_b2, s_ex, Z, lane, false);

    // ---- encoder MLP([ex, h]): 192 -> 128 relu -> 128 (mu | lv) ----
    A[0] = load_A16(s_ex, Z, 0, lane);
    A[1] = load_A16(s_ex, Z, 1, lane);
#pragma unroll
    for (int k = 0; k < 4; ++k) A[2 + k] = load_A16(s_h, H, k, lane);
    mlp_gemm<6, 8>(A, pk + OFF_ENC1, enc_b1, s_hid, H, lane, true);
#pragma unroll
    for (int k = 0; k < 4; ++k) A[k] = load_A16(s_hid, H, k, lane);

#if HAVE_ASYNC_LDS
    wait_async0();  // eps DMA done; overlapped with embx + enc hidden GEMMs
#endif
#pragma unroll 1
    for (int j = 0; j < 4; ++j) {  // mu tile j, lv tile j+4
      v8f dmu = bias_frag(enc_b2, j, lane);
      v8f dlv = bias_frag(enc_b2, j + 4, lane);
#pragma unroll
      for (int k = 0; k < 4; ++k) {
        dmu = wmma16(A[k], load_B16(pk + OFF_ENC2, j * 4 + k, lane), dmu);
        dlv = wmma16(A[k], load_B16(pk + OFF_ENC2, (j + 4) * 4 + k, lane), dlv);
      }
      int col = j * 16 + colb;
#pragma unroll
      for (int r = 0; r < 8; ++r) {
        int m = mb + r;
        float mu = dmu[r], lv = dlv[r];
        float sd = fminf(__expf(0.5f * lv), 100.f);
        s_z[m * Z + col] = (f16)(mu + sd * s_ep[m * Z + col]);
        size_t o = ((size_t)(b0 + m) * T + t) * (2 * Z) + col;
        out_zp[o]     = mu;
        out_zp[o + Z] = lv;
      }
    }

    // ---- prior MLP(h): 128 -> 128 relu -> 128 (pmu | plv) ----
#pragma unroll
    for (int k = 0; k < 4; ++k) A[k] = load_A16(s_h, H, k, lane);
    mlp_gemm<4, 8>(A, pk + OFF_PRI1, prior_b1, s_hid, H, lane, true);
#pragma unroll
    for (int k = 0; k < 4; ++k) A[k] = load_A16(s_hid, H, k, lane);
#pragma unroll 1
    for (int j = 0; j < 8; ++j) {
      v8f d = bias_frag(prior_b2, j, lane);
#pragma unroll
      for (int k = 0; k < 4; ++k)
        d = wmma16(A[k], load_B16(pk + OFF_PRI2, j * 4 + k, lane), d);
      int col = j * 16 + colb;
#pragma unroll
      for (int r = 0; r < 8; ++r)
        out_pp[((size_t)(b0 + mb + r) * T + t) * (2 * Z) + col] = d[r];
    }

    // ---- ez = embz MLP(z): 64 -> 128 relu -> 64 ----
    A[0] = load_A16(s_z, Z, 0, lane);
    A[1] = load_A16(s_z, Z, 1, lane);
    mlp_gemm<2, 8>(A, pk + OFF_EMBZ1, embz_b1, s_hid, H, lane, true);
#pragma unroll
    for (int k = 0; k < 4; ++k) A[k] = load_A16(s_hid, H, k, lane);
    mlp_gemm<4, 4>(A, pk + OFF_EMBZ2, embz_b2, s_ez, Z, lane, false);

    // ---- x_hat = dec MLP([ez, h]): 192 -> 128 relu -> 32 ----
    A[0] = load_A16(s_ez, Z, 0, lane);
    A[1] = load_A16(s_ez, Z, 1, lane);
#pragma unroll
    for (int k = 0; k < 4; ++k) A[2 + k] = load_A16(s_h, H, k, lane);
    mlp_gemm<6, 8>(A, pk + OFF_DEC1, dec_b1, s_hid, H, lane, true);
#pragma unroll
    for (int k = 0; k < 4; ++k) A[k] = load_A16(s_hid, H, k, lane);
#pragma unroll 1
    for (int j = 0; j < 2; ++j) {
      v8f d = bias_frag(dec_b2, j, lane);
#pragma unroll
      for (int k = 0; k < 4; ++k)
        d = wmma16(A[k], load_B16(pk + OFF_DEC2, j * 4 + k, lane), d);
      int col = j * 16 + colb;
#pragma unroll
      for (int r = 0; r < 8; ++r) {
        float v = d[r];
        out_rec[((size_t)(b0 + mb + r) * T + t) * P + col] = v;
        s_xh[(mb + r) * P + col] = (f16)v;
      }
    }

    // ---- exh = embx MLP(x_hat): 32 -> 128 relu -> 64 ----
    A[0] = load_A16(s_xh, P, 0, lane);
    mlp_gemm<1, 8>(A, pk + OFF_EMBX1, embx_b1, s_hid, H, lane, true);
#pragma unroll
    for (int k = 0; k < 4; ++k) A[k] = load_A16(s_hid, H, k, lane);
    mlp_gemm<4, 4>(A, pk + OFF_EMBX2, embx_b2, s_exh, Z, lane, false);

    // ---- LSTM: gates = [exh, ez, h] (K=256) @ [W_ih; W_hh] (N=512) ----
    // A fragments preloaded once; c state lives in LDS (f32).
    {
      v16h AL[8];
      AL[0] = load_A16(s_exh, Z, 0, lane);
      AL[1] = load_A16(s_exh, Z, 1, lane);
      AL[2] = load_A16(s_ez, Z, 0, lane);
      AL[3] = load_A16(s_ez, Z, 1, lane);
#pragma unroll
      for (int k = 0; k < 4; ++k) AL[4 + k] = load_A16(s_h, H, k, lane);
#pragma unroll 1
      for (int j = 0; j < 8; ++j) {  // i/f/g/o share lane/reg positions
        v8f di  = bias2_frag(b_ih, b_hh, j, lane);
        v8f df  = bias2_frag(b_ih, b_hh, j + 8, lane);
        v8f dg  = bias2_frag(b_ih, b_hh, j + 16, lane);
        v8f dob = bias2_frag(b_ih, b_hh, j + 24, lane);
#pragma unroll
        for (int k = 0; k < 8; ++k) {
          const f16* wl = pk + OFF_LSTM;
          di  = wmma16(AL[k], load_B16(wl, (j)*8 + k, lane), di);
          df  = wmma16(AL[k], load_B16(wl, (j + 8) * 8 + k, lane), df);
          dg  = wmma16(AL[k], load_B16(wl, (j + 16) * 8 + k, lane), dg);
          dob = wmma16(AL[k], load_B16(wl, (j + 24) * 8 + k, lane), dob);
        }
        int col = j * 16 + colb;
#pragma unroll
        for (int r = 0; r < 8; ++r) {
          int idx = (mb + r) * H + col;
          float cn = sigmf(df[r]) * s_c[idx] + sigmf(di[r]) * tanhfast(dg[r]);
          s_c[idx] = cn;
          s_h[idx] = (f16)(sigmf(dob[r]) * tanhfast(cn));
        }
      }
    }
  }
}

// ---------------- host launcher ----------------
extern "C" void kernel_launch(void* const* d_in, const int* /*in_sizes*/,
                              int /*n_in*/, void* d_out, int /*out_size*/,
                              void* d_ws, size_t /*ws_size*/,
                              hipStream_t stream) {
  const float* x        = (const float*)d_in[0];
  const float* eps      = (const float*)d_in[1];
  const float* prior_W1 = (const float*)d_in[2];
  const float* prior_b1 = (const float*)d_in[3];
  const float* prior_W2 = (const float*)d_in[4];
  const float* prior_b2 = (const float*)d_in[5];
  const float* embx_W1  = (const float*)d_in[6];
  const float* embx_b1  = (const float*)d_in[7];
  const float* embx_W2  = (const float*)d_in[8];
  const float* embx_b2  = (const float*)d_in[9];
  const float* enc_W1   = (const float*)d_in[10];
  const float* enc_b1   = (const float*)d_in[11];
  const float* enc_W2   = (const float*)d_in[12];
  const float* enc_b2   = (const float*)d_in[13];
  const float* dec_W1   = (const float*)d_in[14];
  const float* dec_b1   = (const float*)d_in[15];
  const float* dec_W2   = (const float*)d_in[16];
  const float* dec_b2   = (const float*)d_in[17];
  const float* embz_W1  = (const float*)d_in[18];
  const float* embz_b1  = (const float*)d_in[19];
  const float* embz_W2  = (const float*)d_in[20];
  const float* embz_b2  = (const float*)d_in[21];
  const float* W_ih     = (const float*)d_in[22];
  const float* W_hh     = (const float*)d_in[23];
  const float* b_ih     = (const float*)d_in[24];
  const float* b_hh     = (const float*)d_in[25];

  f16* pk = (f16*)d_ws;  // 512 KB packed weights
  dim3 pg(256), pb(256);
  //                (W, N, KT_src, NT, kt_base, KT_total, dst)
  pack_kernel<<<pg, pb, 0, stream>>>(embx_W1, 128, 1, 8, 0, 1, pk + OFF_EMBX1);
  pack_kernel<<<pg, pb, 0, stream>>>(embx_W2,  64, 4, 4, 0, 4, pk + OFF_EMBX2);
  pack_kernel<<<pg, pb, 0, stream>>>(enc_W1,  128, 6, 8, 0, 6, pk + OFF_ENC1);
  pack_kernel<<<pg, pb, 0, stream>>>(enc_W2,  128, 4, 8, 0, 4, pk + OFF_ENC2);
  pack_kernel<<<pg, pb, 0, stream>>>(prior_W1,128, 4, 8, 0, 4, pk + OFF_PRI1);
  pack_kernel<<<pg, pb, 0, stream>>>(prior_W2,128, 4, 8, 0, 4, pk + OFF_PRI2);
  pack_kernel<<<pg, pb, 0, stream>>>(embz_W1, 128, 2, 8, 0, 2, pk + OFF_EMBZ1);
  pack_kernel<<<pg, pb, 0, stream>>>(embz_W2,  64, 4, 4, 0, 4, pk + OFF_EMBZ2);
  pack_kernel<<<pg, pb, 0, stream>>>(dec_W1,  128, 6, 8, 0, 6, pk + OFF_DEC1);
  pack_kernel<<<pg, pb, 0, stream>>>(dec_W2,   32, 4, 2, 0, 4, pk + OFF_DEC2);
  pack_kernel<<<pg, pb, 0, stream>>>(W_ih,    512, 4, 32, 0, 8, pk + OFF_LSTM);
  pack_kernel<<<pg, pb, 0, stream>>>(W_hh,    512, 4, 32, 4, 8, pk + OFF_LSTM);

  vrnn_kernel<<<dim3(NWG), dim3(WAVES * 32), 0, stream>>>(
      x, eps, pk, embx_b1, embx_b2, enc_b1, enc_b2, prior_b1, prior_b2,
      embz_b1, embz_b2, dec_b1, dec_b2, b_ih, b_hh, (float*)d_out);
}
